// MAACCritic_90735479095688
// MI455X (gfx1250) — matrix operand order repaired and checked by reading, not compiled, not run on hardware
//
#include <hip/hip_runtime.h>

// ---------------------------------------------------------------------------
// MAAC critic forward on gfx1250 (MI455X).
// All GEMMs: V_WMMA_F32_16X16X32_BF16, f32 accumulate.
// B-operand tiles staged via GLOBAL_LOAD_ASYNC_TO_LDS_B128 (ASYNCcnt) into a
// ping-pong LDS buffer. Wave tile 16x128 (8 n-tiles, 64 acc VGPRs). The last
// k-iteration is peeled so the hot loop is branch-free (stage-next + fixed
// s_wait_asynccnt 4 every iteration). DS-latency bubbles before each WMMA are
// covered by SMT across the 8 waves/block.
// Attention (8x8 per (b,e)) stays in f32 VALU with wave32 shuffles.
// ---------------------------------------------------------------------------

#define AGENTS 8
#define BATCH  16384
#define SDIM   456
#define ACTD   16
#define HDIM   256
#define KP_ENC 480      // 472 padded to 15 k-tiles of 32
#define KT_ENC 15
#define KT_HEAD 8       // K = 256
#define KT_C1  16       // K = 512
#define NT_FULL 16      // N = 256 -> 16 n-tiles
#define NT_WAVE 8       // n-tiles per wave

typedef __attribute__((ext_vector_type(16))) __bf16 v16bf;
typedef __attribute__((ext_vector_type(8)))  float  v8f;

union BF16x16 { uint4 q[2]; v16bf v; };

__device__ __forceinline__ unsigned short f2bf(float f) {
  unsigned int u = __float_as_uint(f);
  u += 0x7FFFu + ((u >> 16) & 1u);          // round-to-nearest-even
  return (unsigned short)(u >> 16);
}
__device__ __forceinline__ float bf2f(unsigned short h) {
  return __uint_as_float(((unsigned int)h) << 16);
}

// async global->LDS copy of one 16B chunk per lane (one wave = 512B)
__device__ __forceinline__ void async_copy_b128(unsigned lds_addr,
                                                const unsigned short* g) {
  asm volatile("global_load_async_to_lds_b128 %0, %1, off"
               :: "v"(lds_addr), "v"((unsigned long long)(size_t)g)
               : "memory");
}

// ---------------------------------------------------------------------------
// Pack concat(states, actions) -> bf16 [A, B, 480] (zero padded cols 472..479)
// ---------------------------------------------------------------------------
__global__ __launch_bounds__(256) void pack_sa_kernel(
    const float* __restrict__ states, const float* __restrict__ actions,
    unsigned short* __restrict__ out) {
  size_t idx = (size_t)blockIdx.x * 256 + threadIdx.x;
  size_t total = (size_t)AGENTS * BATCH * KP_ENC;
  if (idx >= total) return;
  size_t a   = idx / ((size_t)BATCH * KP_ENC);
  size_t rem = idx % ((size_t)BATCH * KP_ENC);
  size_t b   = rem / KP_ENC;
  int    k   = (int)(rem % KP_ENC);
  float v = 0.f;
  if (k < SDIM)             v = states[(a * BATCH + b) * SDIM + k];
  else if (k < SDIM + ACTD) v = actions[(a * BATCH + b) * ACTD + (k - SDIM)];
  out[idx] = f2bf(v);
}

// ---------------------------------------------------------------------------
// Pack a [batch][Kreal x N] f32 weight into WMMA-B fragment order, zero padded
// to KT*32 rows. B-frag layout (16x16x32 bf16): lane l holds col n = l&15,
// K = (l>=16 ? 8:0) + (t&7) + (t>=8 ? 16:0) for t=0..15, packed per (nt, kt) tile:
//   dst[(((bb*NT + nt)*KT + kt)*32 + lane)*16 + t]
// src element address: bb*srcBatchStride + k*sK + (n/nInner)*sNOuter + (n%nInner)
// ---------------------------------------------------------------------------
__global__ __launch_bounds__(256) void pack_w_kernel(
    const float* __restrict__ src, unsigned short* __restrict__ dst,
    int total, size_t srcBatchStride, int Kreal, int KT, int NT,
    int sK, int nInner, int sNOuter) {
  int idx = blockIdx.x * 256 + threadIdx.x;
  if (idx >= total) return;
  int t    = idx & 15;
  int lane = (idx >> 4) & 31;
  int r    = idx >> 9;
  int kt   = r % KT; r /= KT;
  int nt   = r % NT; r /= NT;
  int bb   = r;
  int n    = nt * 16 + (lane & 15);
  int k    = kt * 32 + ((lane >> 4) << 3) + (t & 7) + ((t >> 3) << 4);
  float v  = 0.f;
  if (k < Kreal)
    v = src[bb * srcBatchStride + (size_t)k * sK +
            (size_t)(n / nInner) * sNOuter + (n % nInner)];
  dst[idx] = f2bf(v);
}

// ---------------------------------------------------------------------------
// Generic bf16 GEMM: C[m,n] = act( A[m,:] * B[:,n] + bias[n] ), out bf16.
// M=16384/agent, N=256. Block: 256 thr = 8 waves = 4 m-slices x 2 n-halves;
// block tile 64(M) x 256(N); wave tile 16 x 128.
// ---------------------------------------------------------------------------
__global__ __launch_bounds__(256) void gemm_bf16_wmma(
    const unsigned short* __restrict__ Ab, size_t aStride, int lda,
    const unsigned short* __restrict__ Bp, size_t bStride,
    const float* __restrict__ bias, int biasStride,
    unsigned short* __restrict__ Ob, size_t oStride, int ldc,
    int KT, int doLrelu) {
  __shared__ uint4 ldsB[2][NT_FULL][64];        // ping-pong, 16 n-frags x 1KB

  const int w      = threadIdx.x >> 5;
  const int lane   = threadIdx.x & 31;
  const int mSlice = w & 3;                     // 4 m-slices of 16 rows
  const int nHalf  = w >> 2;                    // frags 0..7 or 8..15
  const int z      = blockIdx.z;

  const unsigned short* Aag = Ab + (size_t)z * aStride;
  const unsigned short* Bag = Bp + (size_t)z * bStride;
  const float* bz = bias ? bias + (size_t)z * biasStride : nullptr;
  unsigned short* Oag = Ob + (size_t)z * oStride;

  const int mBase = blockIdx.x * 64 + mSlice * 16;
  const int mRow  = mBase + (lane & 15);
  const int khalf = (lane >= 16) ? 8 : 0;
  const unsigned short* arowBase = Aag + (size_t)mRow * lda + khalf;

  // wave-relative LDS byte address of the staging buffer
  const unsigned ldsBase = (unsigned)(size_t)(&ldsB[0][0][0]);

  // wave w async-stages n-frags {2w, 2w+1} of k-tile kt (4 ops = 2KB/wave)
  auto stage = [&](int kt, int buf) {
#pragma unroll
    for (int o = 0; o < 4; ++o) {
      int f = w * 2 + (o >> 1);
      int c = (o & 1) * 32 + lane;              // 16B chunk within frag
      const unsigned short* g = Bag + ((size_t)f * KT + kt) * 512 + c * 8;
      unsigned dst = ldsBase + (unsigned)(((buf * NT_FULL + f) * 64 + c) * 16);
      async_copy_b128(dst, g);
    }
  };

  v8f acc[NT_WAVE] = {};

  // one k-tile of math: 16 ds_loads + 8 WMMAs from LDS buffer `buf`
  auto compute = [&](int buf, const BF16x16& a) {
    const uint4* lb = &ldsB[buf][nHalf * NT_WAVE][lane * 2];
    BF16x16 bf[NT_WAVE];
#pragma unroll
    for (int f = 0; f < NT_WAVE; ++f) {
      bf[f].q[0] = lb[f * 64 + 0];
      bf[f].q[1] = lb[f * 64 + 1];
    }
#pragma unroll
    for (int f = 0; f < NT_WAVE; ++f) {
      acc[f] = __builtin_amdgcn_wmma_f32_16x16x32_bf16(
          false, a.v, false, bf[f].v, (short)0, acc[f], false, false);
    }
  };

  BF16x16 aCur, aNext;
  aCur.q[0] = *(const uint4*)(arowBase + 0);
  aCur.q[1] = *(const uint4*)(arowBase + 16);
  aNext = aCur;
  stage(0, 0);

  // branch-free hot loop: always stage k-tile kt+1 while computing kt
  for (int kt = 0; kt < KT - 1; ++kt) {
    stage(kt + 1, (kt + 1) & 1);
    const unsigned short* ar = arowBase + (kt + 1) * 32;
    aNext.q[0] = *(const uint4*)(ar + 0);       // prefetch next A fragment
    aNext.q[1] = *(const uint4*)(ar + 16);
    asm volatile("s_wait_asynccnt 4" ::: "memory");  // own stage(kt) done
    __syncthreads();                 // all waves drained -> k-tile kt visible
    compute(kt & 1, aCur);
    __syncthreads();                 // done reading buf before its reuse
    aCur = aNext;
  }
  // peeled last iteration: nothing left to stage
  asm volatile("s_wait_asynccnt 0" ::: "memory");
  __syncthreads();
  compute((KT - 1) & 1, aCur);

  // epilogue: bias + optional leaky-relu, store bf16
#pragma unroll
  for (int f = 0; f < NT_WAVE; ++f) {
    int n  = (nHalf * NT_WAVE + f) * 16 + (lane & 15);
    float bv = bz ? bz[n] : 0.f;
    int mTop = mBase + ((lane >= 16) ? 8 : 0);
#pragma unroll
    for (int v = 0; v < 8; ++v) {
      float x = acc[f][v] + bv;
      if (doLrelu) x = (x > 0.f) ? x : 0.01f * x;
      Oag[(size_t)(mTop + v) * ldc + n] = f2bf(x);
    }
  }
}

// ---------------------------------------------------------------------------
// Attention: per (b,e) one wave. lane covers d = 2*lane, 2*lane+1.
// logits[i][j] = <sel_i, key_j>/8 (cross-lane shfl reduction), diag = -1e9,
// softmax over j, other[i,d] = sum_j p*val_j[d] -> bf16 into critic_in[:,256:].
// ---------------------------------------------------------------------------
__global__ __launch_bounds__(256) void attn_kernel(
    const unsigned short* __restrict__ keys, const unsigned short* __restrict__ vals,
    const unsigned short* __restrict__ sels, unsigned short* __restrict__ cin) {
  int g    = blockIdx.x * 8 + (threadIdx.x >> 5);
  int lane = threadIdx.x & 31;
  int e = g & 3;
  int b = g >> 2;
  int colBase = e * 64 + lane * 2;

  float k0[AGENTS], k1[AGENTS], v0[AGENTS], v1[AGENTS];
#pragma unroll
  for (int j = 0; j < AGENTS; ++j) {
    unsigned int kw = *(const unsigned int*)(keys + ((size_t)j * BATCH + b) * HDIM + colBase);
    unsigned int vw = *(const unsigned int*)(vals + ((size_t)j * BATCH + b) * HDIM + colBase);
    k0[j] = bf2f((unsigned short)(kw & 0xffff)); k1[j] = bf2f((unsigned short)(kw >> 16));
    v0[j] = bf2f((unsigned short)(vw & 0xffff)); v1[j] = bf2f((unsigned short)(vw >> 16));
  }

  for (int i = 0; i < AGENTS; ++i) {
    unsigned int sw = *(const unsigned int*)(sels + ((size_t)i * BATCH + b) * HDIM + colBase);
    float s0 = bf2f((unsigned short)(sw & 0xffff));
    float s1 = bf2f((unsigned short)(sw >> 16));

    float logit[AGENTS];
#pragma unroll
    for (int j = 0; j < AGENTS; ++j) {
      float p = s0 * k0[j] + s1 * k1[j];
#pragma unroll
      for (int off = 16; off > 0; off >>= 1) p += __shfl_xor(p, off, 32);
      logit[j] = p * 0.125f;                   // 1/sqrt(D=64)
    }
    logit[i] = -1e9f;

    float mx = logit[0];
#pragma unroll
    for (int j = 1; j < AGENTS; ++j) mx = fmaxf(mx, logit[j]);
    float ex[AGENTS], s = 0.f;
#pragma unroll
    for (int j = 0; j < AGENTS; ++j) { ex[j] = __expf(logit[j] - mx); s += ex[j]; }
    float inv = 1.f / s;

    float o0 = 0.f, o1 = 0.f;
#pragma unroll
    for (int j = 0; j < AGENTS; ++j) {
      float pj = ex[j] * inv;
      o0 += pj * v0[j];
      o1 += pj * v1[j];
    }
    unsigned int ow = (unsigned int)f2bf(o0) | ((unsigned int)f2bf(o1) << 16);
    *(unsigned int*)(cin + ((size_t)i * BATCH + b) * 512 + 256 + colBase) = ow;
  }
}

// ---------------------------------------------------------------------------
// Final: all_q = h @ c_W2 + c_b2 (N=16, one WMMA tile per wave), then
// q[m] = all_q[m][argmax(actions[m])]; gather via LDS.
// ---------------------------------------------------------------------------
__global__ __launch_bounds__(256) void critic_out_kernel(
    const unsigned short* __restrict__ h, const unsigned short* __restrict__ W2p,
    const float* __restrict__ b2, const float* __restrict__ actions,
    float* __restrict__ q) {
  __shared__ float ldsQ[128 * 16];

  const int w    = threadIdx.x >> 5;
  const int lane = threadIdx.x & 31;
  const int z    = blockIdx.z;
  const int mBase = blockIdx.x * 128 + w * 16;
  const int mRow  = mBase + (lane & 15);
  const int khalf = (lane >= 16) ? 8 : 0;

  const unsigned short* hag = h + (size_t)z * BATCH * HDIM;
  v8f acc = {};
  for (int kt = 0; kt < KT_HEAD; ++kt) {
    BF16x16 afrag, bfrag;
    const unsigned short* arow = hag + (size_t)mRow * HDIM + kt * 32;
    afrag.q[0] = *(const uint4*)(arow + khalf);
    afrag.q[1] = *(const uint4*)(arow + 16 + khalf);
    const unsigned short* bp = W2p + ((size_t)z * KT_HEAD + kt) * 512 + lane * 16;
    bfrag.q[0] = *(const uint4*)(bp);
    bfrag.q[1] = *(const uint4*)(bp + 8);
    acc = __builtin_amdgcn_wmma_f32_16x16x32_bf16(
        false, afrag.v, false, bfrag.v, (short)0, acc, false, false);
  }

  int n = lane & 15;
  int mLocal = w * 16 + ((lane >= 16) ? 8 : 0);
  float bv = b2[z * ACTD + n];
#pragma unroll
  for (int v = 0; v < 8; ++v) ldsQ[(mLocal + v) * 16 + n] = acc[v] + bv;
  __syncthreads();

  int t = threadIdx.x;
  if (t < 128) {
    int m = blockIdx.x * 128 + t;
    const float* arow = actions + ((size_t)z * BATCH + m) * ACTD;
    float best = arow[0]; int bi = 0;
#pragma unroll
    for (int k = 1; k < ACTD; ++k) {
      float v = arow[k];
      if (v > best) { best = v; bi = k; }      // first-max tie-break (argmax)
    }
    q[(size_t)z * BATCH + m] = ldsQ[t * 16 + bi];
  }
}

// ---------------------------------------------------------------------------
extern "C" void kernel_launch(void* const* d_in, const int* in_sizes, int n_in,
                              void* d_out, int out_size, void* d_ws, size_t ws_size,
                              hipStream_t stream) {
  (void)in_sizes; (void)n_in; (void)out_size; (void)ws_size;
  const float* states  = (const float*)d_in[0];
  const float* actions = (const float*)d_in[1];
  const float* enc_W   = (const float*)d_in[2];
  const float* enc_b   = (const float*)d_in[3];
  const float* s_W     = (const float*)d_in[4];
  const float* s_b     = (const float*)d_in[5];
  const float* key_W   = (const float*)d_in[6];
  const float* sel_W   = (const float*)d_in[7];
  const float* val_W   = (const float*)d_in[8];
  const float* val_b   = (const float*)d_in[9];
  const float* c_W1    = (const float*)d_in[10];
  const float* c_b1    = (const float*)d_in[11];
  const float* c_W2    = (const float*)d_in[12];
  const float* c_b2    = (const float*)d_in[13];
  float* q = (float*)d_out;

  // workspace carve-up (bf16 scratch, 256B aligned)
  char* ws = (char*)d_ws;
  size_t off = 0;
  auto alloc = [&](size_t bytes) -> unsigned short* {
    off = (off + 255) & ~(size_t)255;
    unsigned short* p = (unsigned short*)(ws + off);
    off += bytes;
    return p;
  };
  unsigned short* sa_pad  = alloc((size_t)AGENTS * BATCH * KP_ENC * 2);   // 120 MB
  unsigned short* encWp   = alloc((size_t)AGENTS * NT_FULL * KT_ENC * 512 * 2);
  unsigned short* sWp     = alloc((size_t)AGENTS * NT_FULL * KT_ENC * 512 * 2);
  unsigned short* keyWp   = alloc((size_t)NT_FULL * KT_HEAD * 512 * 2);
  unsigned short* selWp   = alloc((size_t)NT_FULL * KT_HEAD * 512 * 2);
  unsigned short* valWp   = alloc((size_t)NT_FULL * KT_HEAD * 512 * 2);
  unsigned short* c1Wp    = alloc((size_t)AGENTS * NT_FULL * KT_C1 * 512 * 2);
  unsigned short* c2Wp    = alloc((size_t)AGENTS * 1 * KT_HEAD * 512 * 2);
  unsigned short* sa_enc  = alloc((size_t)AGENTS * BATCH * HDIM * 2);     // 64 MB
  unsigned short* cin     = alloc((size_t)AGENTS * BATCH * 512 * 2);      // 128 MB
  unsigned short* keysB   = alloc((size_t)AGENTS * BATCH * HDIM * 2);
  unsigned short* valsB   = alloc((size_t)AGENTS * BATCH * HDIM * 2);
  unsigned short* selsB   = alloc((size_t)AGENTS * BATCH * HDIM * 2);
  unsigned short* hB      = alloc((size_t)AGENTS * BATCH * HDIM * 2);

  // 1) pack activations
  {
    size_t total = (size_t)AGENTS * BATCH * KP_ENC;
    pack_sa_kernel<<<dim3((unsigned)((total + 255) / 256)), 256, 0, stream>>>(
        states, actions, sa_pad);
  }

  // 2) pack weights into WMMA-B fragment order
  auto packW = [&](const float* src, unsigned short* dst, int batch,
                   size_t batchStride, int Kreal, int KT, int NT, int sK,
                   int nInner, int sNOuter) {
    int total = batch * NT * KT * 512;
    pack_w_kernel<<<dim3((total + 255) / 256), 256, 0, stream>>>(
        src, dst, total, batchStride, Kreal, KT, NT, sK, nInner, sNOuter);
  };
  packW(enc_W, encWp, AGENTS, (size_t)(SDIM + ACTD) * HDIM, SDIM + ACTD, KT_ENC, NT_FULL, HDIM, HDIM, 0);
  packW(s_W,   sWp,   AGENTS, (size_t)SDIM * HDIM,          SDIM,        KT_ENC, NT_FULL, HDIM, HDIM, 0);
  packW(key_W, keyWp, 1, 0, HDIM, KT_HEAD, NT_FULL, 64, 64, HDIM * 64);  // [E,H,D] -> n = e*64+d
  packW(sel_W, selWp, 1, 0, HDIM, KT_HEAD, NT_FULL, 64, 64, HDIM * 64);
  packW(val_W, valWp, 1, 0, HDIM, KT_HEAD, NT_FULL, 64, 64, HDIM * 64);
  packW(c_W1,  c1Wp,  AGENTS, (size_t)(2 * HDIM) * HDIM, 2 * HDIM, KT_C1, NT_FULL, HDIM, HDIM, 0);
  packW(c_W2,  c2Wp,  AGENTS, (size_t)HDIM * ACTD, HDIM, KT_HEAD, 1, ACTD, ACTD, 0);

  const dim3 gridG(BATCH / 64, 1, AGENTS);    // block tile 64(M) x 256(N)
  const size_t bpAgentEnc = (size_t)NT_FULL * KT_ENC * 512;
  const size_t bpAgentC1  = (size_t)NT_FULL * KT_C1 * 512;

  // 3) sa_enc = lrelu(sa_pad @ enc_W + enc_b)           [A,B,256] bf16
  gemm_bf16_wmma<<<gridG, 256, 0, stream>>>(
      sa_pad, (size_t)BATCH * KP_ENC, KP_ENC, encWp, bpAgentEnc,
      enc_b, HDIM, sa_enc, (size_t)BATCH * HDIM, HDIM, KT_ENC, 1);

  // 4) s_enc = lrelu(sa_pad @ s_W + s_b) -> critic_in[:, 0:256] (ldc=512)
  gemm_bf16_wmma<<<gridG, 256, 0, stream>>>(
      sa_pad, (size_t)BATCH * KP_ENC, KP_ENC, sWp, bpAgentEnc,
      s_b, HDIM, cin, (size_t)BATCH * 512, 512, KT_ENC, 1);

  // 5) keys = sa_enc @ key_W                              (shared weights)
  gemm_bf16_wmma<<<gridG, 256, 0, stream>>>(
      sa_enc, (size_t)BATCH * HDIM, HDIM, keyWp, 0,
      nullptr, 0, keysB, (size_t)BATCH * HDIM, HDIM, KT_HEAD, 0);

  // 6) vals = lrelu(sa_enc @ val_W + val_b)               (shared bias)
  gemm_bf16_wmma<<<gridG, 256, 0, stream>>>(
      sa_enc, (size_t)BATCH * HDIM, HDIM, valWp, 0,
      val_b, 0, valsB, (size_t)BATCH * HDIM, HDIM, KT_HEAD, 1);

  // 7) sels = s_enc @ sel_W   (A read from critic_in cols 0..255, lda=512)
  gemm_bf16_wmma<<<gridG, 256, 0, stream>>>(
      cin, (size_t)BATCH * 512, 512, selWp, 0,
      nullptr, 0, selsB, (size_t)BATCH * HDIM, HDIM, KT_HEAD, 0);

  // 8) attention -> critic_in[:, 256:512]
  attn_kernel<<<dim3(BATCH * 4 / 8), 256, 0, stream>>>(keysB, valsB, selsB, cin);

  // 9) h = lrelu(critic_in @ c_W1 + c_b1)                 [A,B,256] bf16
  gemm_bf16_wmma<<<gridG, 256, 0, stream>>>(
      cin, (size_t)BATCH * 512, 512, c1Wp, bpAgentC1,
      c_b1, HDIM, hB, (size_t)BATCH * HDIM, HDIM, KT_C1, 1);

  // 10) all_q + argmax gather -> q [A,B]
  critic_out_kernel<<<dim3(BATCH / 128, 1, AGENTS), 256, 0, stream>>>(
      hB, c2Wp, c_b2, actions, q);
}
